// MultiheadSelectiveAttentionWithTokenPruning_56075093016872
// MI455X (gfx1250) — matrix operand order repaired and checked by reading, not compiled, not run on hardware
//
#include <hip/hip_runtime.h>
#include <hip/hip_bf16.h>

#define D_MODEL 1024
#define NHEADS  16
#define DHEAD   64
#define BATCH   4
#define SEQ     2048
#define ROWS    (BATCH * SEQ)   // 8192

typedef __attribute__((ext_vector_type(16))) __bf16 v16bf;
typedef __attribute__((ext_vector_type(8)))  float  v8f;

// ---------------- bf16 helpers (bit-level, RNE) ----------------
__device__ __forceinline__ __bf16 to_bf16(float f) {
  union { float f; unsigned u; } a; a.f = f;
  unsigned r = a.u + 0x7FFFu + ((a.u >> 16) & 1u);
  unsigned short s = (unsigned short)(r >> 16);
  __bf16 h; __builtin_memcpy(&h, &s, 2); return h;
}

// ---------------- WMMA fragment loaders (ISA 7.12.2 layouts) ----------------
// A fragment: 16x32 bf16 tile, rows [row0..row0+15], k [k0..k0+31], row-major src.
__device__ __forceinline__ v16bf load_a(const __bf16* __restrict__ src, int ld,
                                        int row0, int k0) {
  int lane = threadIdx.x & 31;
  int half = lane >> 4;
  const __bf16* p = src + (size_t)(row0 + (lane & 15)) * ld + k0;
  v16bf a;
#pragma unroll
  for (int g = 0; g < 8; ++g) {
    int kb = ((g < 4) ? (2 * g) : (16 + 2 * (g - 4))) + half * 8;
    a[2 * g]     = p[kb];
    a[2 * g + 1] = p[kb + 1];
  }
  return a;
}

// B fragment for C = A * srcNK^T : B[k][n] = srcNK[(n0+n)*ld + k0+k]
// lane holds col n = lane&15 ; element e -> k = (lane>>4)*8 + (e<8 ? e : e+8)
__device__ __forceinline__ v16bf load_bt(const __bf16* __restrict__ src, int ld,
                                         int n0, int k0) {
  int lane = threadIdx.x & 31;
  int half = lane >> 4;
  const __bf16* p = src + (size_t)(n0 + (lane & 15)) * ld + k0 + half * 8;
  v16bf b;
#pragma unroll
  for (int e = 0; e < 16; ++e) b[e] = p[(e < 8) ? e : (e + 8)];
  return b;
}

__device__ __forceinline__ v8f wmma_bf16(v16bf a, v16bf b, v8f c) {
  return __builtin_amdgcn_wmma_f32_16x16x32_bf16(
      /*neg_a=*/false, a, /*neg_b=*/false, b,
      /*c_mod=*/(short)0, c, /*reuse_a=*/false, /*reuse_b=*/false);
}

// ---------------- Kernel 1: fp32 -> bf16 convert ----------------
__global__ void k_cvt(const float* __restrict__ src, __bf16* __restrict__ dst, int n) {
  int i = blockIdx.x * blockDim.x + threadIdx.x;
  if (i < n) dst[i] = to_bf16(src[i]);
}

// ---------------- Kernel 1b: V fp32 -> bf16, transposed per (b,h) ----------------
// V: [B*SEQ][D_MODEL] fp32  ->  Vt: [B][H][DHEAD][SEQ] bf16  (writes coalesced)
__global__ void k_vt(const float* __restrict__ V, __bf16* __restrict__ Vt) {
  int i = blockIdx.x * blockDim.x + threadIdx.x;    // output-linear index
  int n  = i & (SEQ - 1);
  int dh = (i >> 11) & (DHEAD - 1);
  int h  = (i >> 17) & (NHEADS - 1);
  int b  = i >> 21;
  Vt[i] = to_bf16(V[(size_t)(b * SEQ + n) * D_MODEL + h * DHEAD + dh]);
}

// ---------------- Kernel 2: C(MxNc,f32) = A(bf16 MxK) * W(bf16 NcxK)^T ----------------
// One wave per 16x64 strip; A fragment reused across 4 N-tiles.
__global__ void k_gemm_xw(const __bf16* __restrict__ A, const __bf16* __restrict__ W,
                          float* __restrict__ C, int M, int Nc, int K, float scale) {
  int wave = blockIdx.x * (blockDim.x >> 5) + (threadIdx.x >> 5);
  int strips_n = Nc >> 6;
  int m0 = (wave / strips_n) * 16;
  int n0 = (wave % strips_n) * 64;
  if (m0 >= M) return;
  v8f acc[4] = {};
  for (int k0 = 0; k0 < K; k0 += 32) {
    v16bf a = load_a(A, K, m0, k0);
#pragma unroll
    for (int t = 0; t < 4; ++t) {
      v16bf b = load_bt(W, K, n0 + 16 * t, k0);
      acc[t] = wmma_bf16(a, b, acc[t]);
    }
  }
  int lane = threadIdx.x & 31;
  int half = lane >> 4, col = lane & 15;
#pragma unroll
  for (int t = 0; t < 4; ++t)
#pragma unroll
    for (int k = 0; k < 8; ++k)
      C[(size_t)(m0 + half * 8 + k) * Nc + n0 + 16 * t + col] = acc[t][k] * scale;
}

// ---------------- Kernel 3: LayerNorm row (D_MODEL) + bf16 downconvert ----------------
__global__ void k_ln_bf16(const float* __restrict__ src, const float* __restrict__ g,
                          const float* __restrict__ bb, __bf16* __restrict__ dst) {
  int row = blockIdx.x;
  const float* x = src + (size_t)row * D_MODEL;
  __shared__ float red[256];
  float s = 0.f;
  for (int i = threadIdx.x; i < D_MODEL; i += 256) s += x[i];
  red[threadIdx.x] = s; __syncthreads();
  for (int o = 128; o > 0; o >>= 1) {
    if ((int)threadIdx.x < o) red[threadIdx.x] += red[threadIdx.x + o];
    __syncthreads();
  }
  float mean = red[0] * (1.0f / D_MODEL);
  __syncthreads();
  float v = 0.f;
  for (int i = threadIdx.x; i < D_MODEL; i += 256) { float d = x[i] - mean; v += d * d; }
  red[threadIdx.x] = v; __syncthreads();
  for (int o = 128; o > 0; o >>= 1) {
    if ((int)threadIdx.x < o) red[threadIdx.x] += red[threadIdx.x + o];
    __syncthreads();
  }
  float rstd = rsqrtf(red[0] * (1.0f / D_MODEL) + 1e-5f);
  for (int i = threadIdx.x; i < D_MODEL; i += 256)
    dst[(size_t)row * D_MODEL + i] = to_bf16((x[i] - mean) * rstd * g[i] + bb[i]);
}

// ---------------- Kernel 4: head-0 logits L[b,n,m] = (Q0[n]·K0[m])/8 -> F region ----------------
__global__ void k_logits0(const __bf16* __restrict__ Qb, const __bf16* __restrict__ Kb,
                          float* __restrict__ F) {
  int wave = blockIdx.x * (blockDim.x >> 5) + (threadIdx.x >> 5);
  const int strips_n = SEQ >> 6;                 // 32
  const int per_b = (SEQ >> 4) * strips_n;       // 4096
  int b = wave / per_b;
  int r = wave % per_b;
  int m0 = (r / strips_n) * 16;
  int n0 = (r % strips_n) * 64;
  const __bf16* Q = Qb + (size_t)b * SEQ * D_MODEL;   // head 0 = cols [0,64)
  const __bf16* Kp = Kb + (size_t)b * SEQ * D_MODEL;
  float* Fb = F + (size_t)b * SEQ * SEQ;
  v8f acc[4] = {};
#pragma unroll
  for (int t = 0; t < 4; ++t) {
    acc[t] = wmma_bf16(load_a(Q, D_MODEL, m0, 0),  load_bt(Kp, D_MODEL, n0 + 16 * t, 0),  acc[t]);
    acc[t] = wmma_bf16(load_a(Q, D_MODEL, m0, 32), load_bt(Kp, D_MODEL, n0 + 16 * t, 32), acc[t]);
  }
  int lane = threadIdx.x & 31;
  int half = lane >> 4, col = lane & 15;
#pragma unroll
  for (int t = 0; t < 4; ++t)
#pragma unroll
    for (int k = 0; k < 8; ++k)
      Fb[(size_t)(m0 + half * 8 + k) * SEQ + n0 + 16 * t + col] = acc[t][k] * 0.125f;
}

// ---------------- Kernel 5: in-place causal relu cumsum -> F_mask ----------------
// F[n,m] = sum_{i=1..n} relu(L[i-1,m]) restricted to 1 <= m <= i-2
// (reproduces zero-col0, zero-diag, roll-down, zero-row0, cumsum of the reference)
__global__ void k_fmask(float* __restrict__ F) {
  int m = blockIdx.x * blockDim.x + threadIdx.x;
  int b = blockIdx.y;
  float* Fb = F + (size_t)b * SEQ * SEQ;
  float acc = 0.f, prev = 0.f;
  for (int n = 0; n < SEQ; ++n) {
    float cur = Fb[(size_t)n * SEQ + m];          // read L before overwrite
    if (n >= 1 && m >= 1 && m <= n - 2) acc += fmaxf(prev, 0.f);
    Fb[(size_t)n * SEQ + m] = acc;
    prev = cur;
  }
}

// ---------------- Kernel 6: flash-style attention with F_mask ----------------
// One wave per (b, h, 16-query block); 32 keys per iteration; online softmax.
__global__ void k_attn(const __bf16* __restrict__ Qb, const __bf16* __restrict__ Kb,
                       const __bf16* __restrict__ Vt, const float* __restrict__ F,
                       __bf16* __restrict__ Ob) {
  __shared__ __bf16 plds[4][16][32];
  int wv = threadIdx.x >> 5;
  int lane = threadIdx.x & 31;
  int wave = blockIdx.x * 4 + wv;
  const int qblks = SEQ >> 4;                    // 128
  int b = wave / (NHEADS * qblks);
  int rem = wave % (NHEADS * qblks);
  int h = rem / qblks;
  int q0 = (rem % qblks) * 16;
  const __bf16* Q  = Qb + (size_t)b * SEQ * D_MODEL + h * DHEAD;
  const __bf16* Kp = Kb + (size_t)b * SEQ * D_MODEL + h * DHEAD;
  const __bf16* Vp = Vt + (size_t)(b * NHEADS + h) * DHEAD * SEQ;  // [DHEAD][SEQ]
  const float* Fb  = F  + (size_t)b * SEQ * SEQ;
  int half = lane >> 4, col = lane & 15;

  float mrow[8], lrow[8];
#pragma unroll
  for (int k = 0; k < 8; ++k) { mrow[k] = -3.0e38f; lrow[k] = 0.f; }
  v8f o[4] = {};

  v16bf qa0 = load_a(Q, D_MODEL, q0, 0);
  v16bf qa1 = load_a(Q, D_MODEL, q0, 32);

  for (int j = 0; j <= q0 + 15; j += 32) {
    v8f s1 = {}, s2 = {};
    s1 = wmma_bf16(qa0, load_bt(Kp, D_MODEL, j, 0),       s1);
    s1 = wmma_bf16(qa1, load_bt(Kp, D_MODEL, j, 32),      s1);
    s2 = wmma_bf16(qa0, load_bt(Kp, D_MODEL, j + 16, 0),  s2);
    s2 = wmma_bf16(qa1, load_bt(Kp, D_MODEL, j + 16, 32), s2);

#pragma unroll
    for (int k = 0; k < 8; ++k) {
      int row = q0 + half * 8 + k;
      int c1 = j + col, c2 = j + 16 + col;
      float v1 = (c1 <= row) ? s1[k] * 0.125f - Fb[(size_t)row * SEQ + c1] : -3.0e38f;
      float v2 = (c2 <= row) ? s2[k] * 0.125f - Fb[(size_t)row * SEQ + c2] : -3.0e38f;
      float mx = fmaxf(v1, v2);
#pragma unroll
      for (int off = 1; off < 16; off <<= 1) mx = fmaxf(mx, __shfl_xor(mx, off, 16));
      float mnew = fmaxf(mrow[k], mx);
      float p1 = __expf(v1 - mnew);
      float p2 = __expf(v2 - mnew);
      float ps = p1 + p2;
#pragma unroll
      for (int off = 1; off < 16; off <<= 1) ps += __shfl_xor(ps, off, 16);
      float corr = __expf(mrow[k] - mnew);
      lrow[k] = lrow[k] * corr + ps;
      mrow[k] = mnew;
#pragma unroll
      for (int t = 0; t < 4; ++t) o[t][k] *= corr;   // online rescale
      plds[wv][half * 8 + k][col]      = to_bf16(p1);
      plds[wv][half * 8 + k][col + 16] = to_bf16(p2);
    }
    // same-wave LDS is in-order; pin compiler + HW ordering anyway
    __builtin_amdgcn_wave_barrier();
    asm volatile("s_wait_dscnt 0" ::: "memory");

    // re-read probs as an A fragment (transpose through LDS)
    v16bf pa;
    {
      int rowA = lane & 15;
#pragma unroll
      for (int g = 0; g < 8; ++g) {
        int kb = ((g < 4) ? (2 * g) : (16 + 2 * (g - 4))) + half * 8;
        pa[2 * g]     = plds[wv][rowA][kb];
        pa[2 * g + 1] = plds[wv][rowA][kb + 1];
      }
    }
    __builtin_amdgcn_wave_barrier();
#pragma unroll
    for (int t = 0; t < 4; ++t) {
      // B[k][n] = Vt[n][j+k]  (two runs of 8 keys per lane, table order)
      v16bf vb = load_bt(Vp, SEQ, t * 16, j);
      o[t] = wmma_bf16(pa, vb, o[t]);
    }
  }
  // normalize + store bf16 O
#pragma unroll
  for (int t = 0; t < 4; ++t)
#pragma unroll
    for (int k = 0; k < 8; ++k) {
      int row = q0 + half * 8 + k;
      Ob[(size_t)(b * SEQ + row) * D_MODEL + h * DHEAD + t * 16 + col] =
          to_bf16(o[t][k] / lrow[k]);
    }
}

// ---------------- host-side launch ----------------
extern "C" void kernel_launch(void* const* d_in, const int* in_sizes, int n_in,
                              void* d_out, int out_size, void* d_ws, size_t ws_size,
                              hipStream_t stream) {
  (void)in_sizes; (void)n_in; (void)out_size; (void)ws_size;
  const float* X  = (const float*)d_in[0];
  const float* Wq = (const float*)d_in[1];
  const float* Wk = (const float*)d_in[2];
  const float* Wv = (const float*)d_in[3];
  const float* Wo = (const float*)d_in[4];
  const float* gq = (const float*)d_in[5];
  const float* bq = (const float*)d_in[6];
  const float* gk = (const float*)d_in[7];
  const float* bk = (const float*)d_in[8];
  // d_in[9] = start_pos (unused, ==0)

  float* out0 = (float*)d_out;                         // (B,N,D_MODEL)
  float* Fm   = out0 + (size_t)ROWS * D_MODEL;         // (B,N,N) F_mask

  // workspace layout (bytes)
  char* ws = (char*)d_ws;
  __bf16* Xb  = (__bf16*)(ws);                              // 16 MB
  __bf16* Wqb = (__bf16*)(ws + 16777216);                   // 2 MB each
  __bf16* Wkb = (__bf16*)(ws + 18874368);
  __bf16* Wvb = (__bf16*)(ws + 20971520);
  __bf16* Wob = (__bf16*)(ws + 23068672);
  float*  LIN = (float*) (ws + 25165824);                   // 32 MB (reused Q/K/V)
  __bf16* Qbn = (__bf16*)(ws + 58720256);                   // 16 MB each
  __bf16* Kbn = (__bf16*)(ws + 75497472);
  __bf16* Vtb = (__bf16*)(ws + 92274688);                   // transposed V
  __bf16* Obn = (__bf16*)(ws + 109051904);                  // end ~120 MB

  const int nX = ROWS * D_MODEL;          // 8388608
  const int nW = D_MODEL * D_MODEL;       // 1048576

  // 1) convert inputs to bf16
  k_cvt<<<nX / 256, 256, 0, stream>>>(X,  Xb,  nX);
  k_cvt<<<nW / 256, 256, 0, stream>>>(Wq, Wqb, nW);
  k_cvt<<<nW / 256, 256, 0, stream>>>(Wk, Wkb, nW);
  k_cvt<<<nW / 256, 256, 0, stream>>>(Wv, Wvb, nW);
  k_cvt<<<nW / 256, 256, 0, stream>>>(Wo, Wob, nW);

  const int gemm_waves = (ROWS / 16) * (D_MODEL / 64);   // 8192
  const int gemm_blocks = gemm_waves / 4;                // 2048

  // 2) Q = LN(X Wq^T) ; K = LN(X Wk^T) ; V = X Wv^T   (LIN reused sequentially)
  k_gemm_xw<<<gemm_blocks, 128, 0, stream>>>(Xb, Wqb, LIN, ROWS, D_MODEL, D_MODEL, 1.0f);
  k_ln_bf16<<<ROWS, 256, 0, stream>>>(LIN, gq, bq, Qbn);
  k_gemm_xw<<<gemm_blocks, 128, 0, stream>>>(Xb, Wkb, LIN, ROWS, D_MODEL, D_MODEL, 1.0f);
  k_ln_bf16<<<ROWS, 256, 0, stream>>>(LIN, gk, bk, Kbn);
  k_gemm_xw<<<gemm_blocks, 128, 0, stream>>>(Xb, Wvb, LIN, ROWS, D_MODEL, D_MODEL, 1.0f);
  k_vt<<<nX / 256, 256, 0, stream>>>(LIN, Vtb);

  // 3) head-0 logits directly into F region, then in-place masked cumsum
  const int l0_waves = BATCH * (SEQ / 16) * (SEQ / 64);  // 16384
  k_logits0<<<l0_waves / 4, 128, 0, stream>>>(Qbn, Kbn, Fm);
  dim3 fgrid(SEQ / 256, BATCH);
  k_fmask<<<fgrid, 256, 0, stream>>>(Fm);

  // 4) attention -> bf16 O
  const int attn_waves = BATCH * NHEADS * (SEQ / 16);    // 8192
  k_attn<<<attn_waves / 4, 128, 0, stream>>>(Qbn, Kbn, Vtb, Fm, Obn);

  // 5) output projection -> d_out
  k_gemm_xw<<<gemm_blocks, 128, 0, stream>>>(Obn, Wob, out0, ROWS, D_MODEL, D_MODEL, 1.0f);
}
